// VirtualCE_21337397527201
// MI455X (gfx1250) — matrix operand order repaired and checked by reading, not compiled, not run on hardware
//
#include <hip/hip_runtime.h>
#include <hip/hip_bf16.h>

typedef _Float16 v16h __attribute__((ext_vector_type(16)));
typedef _Float16 v8h  __attribute__((ext_vector_type(8)));
typedef _Float16 v4h  __attribute__((ext_vector_type(4)));
typedef float    v8f  __attribute__((ext_vector_type(8)));

#define NROW 262144
#define D    128
#define C    512
#define NSLOT 16
#define INV_BETA 10.0f

// workspace layout in float units
#define CSUM_OFF 0
#define CSUM_N   (NSLOT * C * D)            // 16 replicated [C][D] f32 accumulators
#define CNT_OFF  (CSUM_OFF + CSUM_N)
#define CNT_N    (NSLOT * C)
#define LOSS_OFF (CNT_OFF + CNT_N)
#define PRES_OFF (LOSS_OFF + 16)
#define CH_OFF   (PRES_OFF + C)             // C*D halves  (f16 centers)
#define XH_OFF   (CH_OFF + (C * D) / 2)     // NROW*D halves (f16 normalized features)
#define ZERO_N   (PRES_OFF)                 // zero csum + counts + loss

union V16 { v16h v; v8h h[2]; };

__device__ __forceinline__ float wave_red_add(float v) {
    #pragma unroll
    for (int o = 16; o > 0; o >>= 1) v += __shfl_xor(v, o, 32);
    return v;
}

// ---------------- K0: zero accumulators ----------------
__global__ void vce_k0_zero(float* wsf) {
    int i = blockIdx.x * blockDim.x + threadIdx.x;
    if (i < ZERO_N) wsf[i] = 0.0f;
}

// ---------------- K1: normalize rows, emit f16 copy, scatter class sums ----------------
__global__ void vce_k1_norm_scatter(const float* __restrict__ x,
                                    const int* __restrict__ tgt,
                                    float* wsf) {
    const int wave = threadIdx.x >> 5;
    const int lane = threadIdx.x & 31;
    const int row  = blockIdx.x * (blockDim.x >> 5) + wave;

    const float4* rp = (const float4*)(x + (size_t)row * D);
    float4 v = rp[lane];                               // 4 dims per lane, 128/row
    float ss = v.x * v.x + v.y * v.y + v.z * v.z + v.w * v.w;
    ss = wave_red_add(ss);
    float r = 1.0f / fmaxf(sqrtf(ss), 1e-12f);
    float y0 = v.x * r, y1 = v.y * r, y2 = v.z * r, y3 = v.w * r;

    _Float16* xh = (_Float16*)(wsf + XH_OFF);
    v4h o = { (_Float16)y0, (_Float16)y1, (_Float16)y2, (_Float16)y3 };
    *(v4h*)(xh + (size_t)row * D + lane * 4) = o;

    const int t    = tgt[row];
    const int slot = blockIdx.x & (NSLOT - 1);
    float* cs = wsf + CSUM_OFF + ((size_t)slot * C + t) * D + lane * 4;
    __hip_atomic_fetch_add(cs + 0, y0, __ATOMIC_RELAXED, __HIP_MEMORY_SCOPE_AGENT);
    __hip_atomic_fetch_add(cs + 1, y1, __ATOMIC_RELAXED, __HIP_MEMORY_SCOPE_AGENT);
    __hip_atomic_fetch_add(cs + 2, y2, __ATOMIC_RELAXED, __HIP_MEMORY_SCOPE_AGENT);
    __hip_atomic_fetch_add(cs + 3, y3, __ATOMIC_RELAXED, __HIP_MEMORY_SCOPE_AGENT);
    if (lane == 0)
        __hip_atomic_fetch_add(wsf + CNT_OFF + slot * C + t, 1.0f,
                               __ATOMIC_RELAXED, __HIP_MEMORY_SCOPE_AGENT);
}

// ---------------- K2: reduce slots -> centers (f16) + present mask ----------------
__global__ void vce_k2_centers(float* wsf) {
    const int wave = threadIdx.x >> 5;
    const int lane = threadIdx.x & 31;
    const int c    = blockIdx.x * (blockDim.x >> 5) + wave;
    if (c >= C) return;

    float4 s = { 0.f, 0.f, 0.f, 0.f };
    float cnt = 0.f;
    #pragma unroll
    for (int sl = 0; sl < NSLOT; ++sl) {
        const float4* p = (const float4*)(wsf + CSUM_OFF + ((size_t)sl * C + c) * D);
        float4 q = p[lane];
        s.x += q.x; s.y += q.y; s.z += q.z; s.w += q.w;
        cnt += wsf[CNT_OFF + sl * C + c];
    }
    float inv_cnt = 1.0f / fmaxf(cnt, 1.0f);
    float m0 = s.x * inv_cnt, m1 = s.y * inv_cnt, m2 = s.z * inv_cnt, m3 = s.w * inv_cnt;
    float ss = m0 * m0 + m1 * m1 + m2 * m2 + m3 * m3;
    ss = wave_red_add(ss);
    float r = 1.0f / fmaxf(sqrtf(ss), 1e-12f);

    _Float16* ch = (_Float16*)(wsf + CH_OFF);
    v4h o = { (_Float16)(m0 * r), (_Float16)(m1 * r), (_Float16)(m2 * r), (_Float16)(m3 * r) };
    *(v4h*)(ch + (size_t)c * D + lane * 4) = o;

    if (lane == 0) wsf[PRES_OFF + c] = (cnt > 0.f) ? 1.0f : 0.0f;
    if (blockIdx.x == 0 && threadIdx.x == 0) wsf[LOSS_OFF] = 0.0f;
}

// ---------------- K3: WMMA logits + softmax-CE reduction ----------------
// One wave handles 16 samples x all 512 classes: 32 class-tiles, each
// 16x16x128 = 4x v_wmma_f32_16x16x32_f16.  A-tile loads are hoisted into one
// clause ahead of the 4 back-to-back WMMAs so the scheduler can overlap the
// next tile's loads with the exp/mask epilogue of the current tile.
__global__ void vce_k3_wmma_ce(const int* __restrict__ tgt, float* wsf, int n) {
    const int wid  = blockIdx.x * (blockDim.x >> 5) + (threadIdx.x >> 5);
    const int lane = threadIdx.x & 31;
    const int col  = lane & 15;       // N index (sample within tile)
    const int half = lane >> 4;       // lane half selects K sub-range
    const int s0   = wid * 16;

    const _Float16* xh   = (const _Float16*)(wsf + XH_OFF);
    const _Float16* ch   = (const _Float16*)(wsf + CH_OFF);
    const float*    pres = wsf + PRES_OFF;

    // B (32x16, KxN): lane = column(sample); lanes 0-15 K {0..7,16..23}, 16-31 K {8..15,24..31}
    const _Float16* xrow = xh + (size_t)(s0 + col) * D;
    V16 B[4];
    #pragma unroll
    for (int kb = 0; kb < 4; ++kb) {
        B[kb].h[0] = *(const v8h*)(xrow + kb * 32 + half * 8);
        B[kb].h[1] = *(const v8h*)(xrow + kb * 32 + 16 + half * 8);
    }
    const int t = tgt[s0 + col];

    float dpart = 0.0f;   // partial denominator over this lane-half's class rows
    float ppart = 0.0f;   // own-class logit (scaled)

    for (int ct = 0; ct < 32; ++ct) {
        // A (16x32, MxK): lane = row(class); lanes 0-15 K {0..7,16..23}, 16-31 K {8..15,24..31}
        const _Float16* crow = ch + (size_t)(ct * 16 + col) * D;

        // (a) load the whole 16x128 A tile in one clause
        V16 A[4];
        #pragma unroll
        for (int kb = 0; kb < 4; ++kb) {
            A[kb].h[0] = *(const v8h*)(crow + kb * 32 + half * 8);
            A[kb].h[1] = *(const v8h*)(crow + kb * 32 + 16 + half * 8);
        }

        // (b) 4 back-to-back WMMAs
        v8f acc = {};
        #pragma unroll
        for (int kb = 0; kb < 4; ++kb) {
            acc = __builtin_amdgcn_wmma_f32_16x16x32_f16(
                false, A[kb].v, false, B[kb].v, (short)0, acc, false, false);
        }

        // (c) epilogue: masked exp accumulation + own-class logit pick
        const int cb = ct * 16 + 8 * half;   // D layout: M = r + 8*half
        #pragma unroll
        for (int r = 0; r < 8; ++r) {
            float l = acc[r] * INV_BETA;
            dpart += pres[cb + r] * __expf(l);
            ppart += (cb + r == t) ? l : 0.0f;
        }
    }

    // combine lane halves: full denom + pos per sample column
    float denom = dpart + __shfl_xor(dpart, 16, 32);
    float pos   = ppart + __shfl_xor(ppart, 16, 32);
    float contrib = pos - __logf(denom);
    #pragma unroll
    for (int o = 1; o < 16; o <<= 1) contrib += __shfl_xor(contrib, o, 32);
    if (lane == 0)
        __hip_atomic_fetch_add(wsf + LOSS_OFF, contrib,
                               __ATOMIC_RELAXED, __HIP_MEMORY_SCOPE_AGENT);
    (void)n;
}

// ---------------- K4: finalize ----------------
__global__ void vce_k4_final(const float* wsf, float* out, float inv_n) {
    if (threadIdx.x == 0 && blockIdx.x == 0)
        out[0] = -wsf[LOSS_OFF] * inv_n;
}

extern "C" void kernel_launch(void* const* d_in, const int* in_sizes, int n_in,
                              void* d_out, int out_size, void* d_ws, size_t ws_size,
                              hipStream_t stream) {
    const float* x   = (const float*)d_in[0];
    const int*   tgt = (const int*)d_in[1];
    float*       wsf = (float*)d_ws;
    float*       out = (float*)d_out;

    const int n = in_sizes[1];            // 262144 samples (d=128, C=512 fixed)

    // K0: zero class-sum slots, counts, loss
    vce_k0_zero<<<(ZERO_N + 255) / 256, 256, 0, stream>>>(wsf);

    // K1: one wave per row; 8 rows per 256-thread block
    vce_k1_norm_scatter<<<n / 8, 256, 0, stream>>>(x, tgt, wsf);

    // K2: one wave per class; 4 classes per 128-thread block
    vce_k2_centers<<<C / 4, 128, 0, stream>>>(wsf);

    // K3: one wave per 16 samples; 8 waves per 256-thread block
    vce_k3_wmma_ce<<<n / 128, 256, 0, stream>>>(tgt, wsf, n);

    // K4: scalar finalize
    vce_k4_final<<<1, 32, 0, stream>>>(wsf, out, 1.0f / (float)n);
}
